// MambaResBlock_68032281968823
// MI455X (gfx1250) — matrix operand order repaired and checked by reading, not compiled, not run on hardware
//
#include <hip/hip_runtime.h>
#include <hip/hip_bf16.h>
#include <cstdint>

typedef __attribute__((ext_vector_type(16))) __bf16 v16bf;
typedef __attribute__((ext_vector_type(8)))  float  v8f;

#define BN   4
#define NF_  64
#define DST  16
#define DIN  128
#define DTR  4
#define LSEQ 9216              // 96*96
#define MTOK (BN*LSEQ)         // 36864 token rows
#define NCH  96                // scan chunks per sequence
#define LCH  (LSEQ/NCH)        // 96 steps per chunk

__device__ __forceinline__ unsigned short f2bf_bits(float f) {
  union { float f; unsigned int u; } v; v.f = f;
  unsigned int r = v.u + 0x7FFFu + ((v.u >> 16) & 1u);   // RNE
  return (unsigned short)(r >> 16);
}
__device__ __forceinline__ float sigmoidf_(float v) { return 1.f / (1.f + __expf(-v)); }

// ---------------- f32 weights -> bf16 (zero-pad tail rows) ---------------------
__global__ __launch_bounds__(256)
void k_w2bf(const float* __restrict__ src, unsigned short* __restrict__ dst,
            int nvalid, int ntot) {
  int tid = blockIdx.x * 256 + threadIdx.x;
  if (tid >= ntot) return;
  dst[tid] = (tid < nvalid) ? f2bf_bits(src[tid]) : (unsigned short)0;
}

// ---------------- (B,C,L) f32 -> token-major (M,C) bf16, LDS-tiled transpose ----
__global__ __launch_bounds__(256)
void k_x_to_tokens(const float* __restrict__ x, unsigned short* __restrict__ xt) {
  __shared__ float tile[64][65];
  int b  = blockIdx.x / (LSEQ / 64);
  int lt = blockIdx.x % (LSEQ / 64);
  for (int e = threadIdx.x; e < 64 * 64; e += 256) {
    int c = e >> 6, li = e & 63;                               // coalesced over li
    tile[c][li] = x[((size_t)(b * NF_ + c)) * LSEQ + lt * 64 + li];
  }
  __syncthreads();
  for (int e = threadIdx.x; e < 64 * 64; e += 256) {
    int li = e >> 6, c = e & 63;                               // coalesced over c
    xt[((size_t)(b * LSEQ + lt * 64 + li)) * NF_ + c] = f2bf_bits(tile[c][li]);
  }
}

// ---------------- GEMM: out[m,n] = sum_k A_bf16[m,k] * Wbf16[n,k] (+bias) ------
// One wave computes a 16 x (16*NT) strip: A fragment reused across NT WMMA tiles.
// Weight buffer is pre-converted bf16 and zero-padded to Npad rows -> no EXEC
// divergence anywhere before the WMMAs.
template <bool OUT_BF16, int NT>
__global__ __launch_bounds__(256)
void k_gemm_wmma(const unsigned short* __restrict__ A,
                 const unsigned short* __restrict__ Wb,
                 const float* __restrict__ bias, void* __restrict__ Out,
                 int M, int Ngroups, int Nvalid, int K, int ldOut) {
  int wave    = blockIdx.x * (blockDim.x >> 5) + (threadIdx.x >> 5);
  int totTile = (M >> 4) * Ngroups;
  if (wave >= totTile) return;                 // wave-uniform: EXEC stays all-1s
  int tm = wave / Ngroups, tg = wave % Ngroups;
  int tn0  = tg * NT;
  int lane = threadIdx.x & 31;
  int half = lane >> 4, r = lane & 15;
  const unsigned short* arow = A + (size_t)(tm * 16 + r) * K;
  const unsigned short* wrow[NT];
#pragma unroll
  for (int i = 0; i < NT; i++)
    wrow[i] = Wb + (size_t)((tn0 + i) * 16 + r) * K;

  v8f acc[NT];
#pragma unroll
  for (int i = 0; i < NT; i++) acc[i] = (v8f){0.f,0.f,0.f,0.f,0.f,0.f,0.f,0.f};

  for (int k0 = 0; k0 < K; k0 += 32) {
    // A fragment: lanes 0-15 hold K{0..7,16..23}, lanes 16-31 hold K{8..15,24..31}
    union { uint4 q[2]; v16bf v; } au;
    au.q[0] = *(const uint4*)(arow + k0 + half * 8);
    au.q[1] = *(const uint4*)(arow + k0 + 16 + half * 8);
    // B fragments: column n = row n of Wb; lanes 0-15 K{0..7,16..23}, lanes 16-31 K{8..15,24..31}
    union { uint4 q[2]; v16bf v; } bu[NT];
#pragma unroll
    for (int i = 0; i < NT; i++) {
      bu[i].q[0] = *(const uint4*)(wrow[i] + k0 + half * 8);
      bu[i].q[1] = *(const uint4*)(wrow[i] + k0 + 16 + half * 8);
    }
#pragma unroll
    for (int i = 0; i < NT; i++)
      acc[i] = __builtin_amdgcn_wmma_f32_16x16x32_bf16(false, au.v, false, bu[i].v,
                                                       (short)0, acc[i], false, false);
  }
#pragma unroll
  for (int i = 0; i < NT; i++) {
    int col = (tn0 + i) * 16 + r;
    if (col < Nvalid) {
      float bb = bias ? bias[col] : 0.f;
#pragma unroll
      for (int e = 0; e < 8; e++) {
        int row = tm * 16 + e + 8 * half;      // C/D layout: VGPR e -> M=e / M=e+8
        float v = acc[i][e] + bb;
        if (OUT_BF16) ((unsigned short*)Out)[(size_t)row * ldOut + col] = f2bf_bits(v);
        else          ((float*)Out)[(size_t)row * ldOut + col] = v;
      }
    }
  }
}

// ---------------- depthwise causal conv1d (k=4) + SiLU; emit f32 + bf16 --------
__global__ __launch_bounds__(256)
void k_dwconv_silu(const float* __restrict__ xz, const float* __restrict__ w1d,
                   const float* __restrict__ b1d, float* __restrict__ xcs,
                   unsigned short* __restrict__ xcsb) {
  int tid = blockIdx.x * 256 + threadIdx.x;
  if (tid >= MTOK * DIN) return;
  int d = tid & (DIN - 1);
  int m = tid >> 7;
  int b = m / LSEQ, l = m % LSEQ;
  float acc = b1d[d];
#pragma unroll
  for (int j = 0; j < 4; j++) {
    int li = l - 3 + j;
    if (li >= 0) acc += xz[((size_t)(b * LSEQ + li)) * (2 * DIN) + d] * w1d[d * 4 + j];
  }
  float s = acc * sigmoidf_(acc);
  xcs[tid]  = s;
  xcsb[tid] = f2bf_bits(s);
}

// ---------------- delta = softplus(dt @ dt_proj_w.T + b), K=4 scalar ------------
__global__ __launch_bounds__(256)
void k_delta(const float* __restrict__ dbc, const float* __restrict__ dtw,
             const float* __restrict__ dtb, float* __restrict__ delta) {
  int tid = blockIdx.x * 256 + threadIdx.x;
  if (tid >= MTOK * DIN) return;
  int d = tid & (DIN - 1);
  int m = tid >> 7;
  const float* dr = dbc + (size_t)m * 36;
  float s = dtb[d];
#pragma unroll
  for (int r2 = 0; r2 < DTR; r2++) s += dr[r2] * dtw[d * DTR + r2];
  delta[tid] = (s > 20.f) ? s : log1pf(__expf(s));
}

// ---------------- scan pass A: per-chunk (prod dA, local q) --------------------
__global__ __launch_bounds__(128)
void k_scan_local(const float* __restrict__ delta, const float* __restrict__ u,
                  const float* __restrict__ dbc, const float* __restrict__ A_log,
                  float* __restrict__ Pq) {
  int b = blockIdx.x / NCH, ch = blockIdx.x % NCH;
  int d = threadIdx.x;
  float Ar[DST];
#pragma unroll
  for (int n = 0; n < DST; n++) Ar[n] = -__expf(A_log[d * DST + n]);
  float P[DST], q[DST];
#pragma unroll
  for (int n = 0; n < DST; n++) { P[n] = 1.f; q[n] = 0.f; }
  int l0 = ch * LCH;
  for (int t = 0; t < LCH; t++) {
    size_t m = (size_t)b * LSEQ + l0 + t;
    float de = delta[m * DIN + d];
    float uu = u[m * DIN + d];
    __builtin_prefetch(&delta[(m + 1) * DIN + d], 0, 1);
    __builtin_prefetch(&u[(m + 1) * DIN + d], 0, 1);
    const float* bp = dbc + m * 36 + DTR;
    float du = de * uu;
#pragma unroll
    for (int n = 0; n < DST; n++) {
      float dA = __expf(de * Ar[n]);
      P[n] *= dA;
      q[n] = dA * q[n] + du * bp[n];
    }
  }
  size_t idx = ((size_t)blockIdx.x * DIN + d) * 32;
#pragma unroll
  for (int n = 0; n < DST; n++) { Pq[idx + n] = P[n]; Pq[idx + 16 + n] = q[n]; }
}

// ---------------- scan pass B: prefix over chunks (parallel in b,d,n) ----------
__global__ __launch_bounds__(256)
void k_scan_prefix(const float* __restrict__ Pq, float* __restrict__ h0) {
  int tid = blockIdx.x * 256 + threadIdx.x;
  if (tid >= BN * DIN * DST) return;
  int b = tid / (DIN * DST);
  int rem = tid % (DIN * DST);
  int d = rem / DST, n = rem % DST;
  float h = 0.f;
  for (int ch = 0; ch < NCH; ch++) {
    size_t idx = (size_t)(b * NCH + ch) * DIN + d;
    h0[idx * DST + n] = h;
    h = Pq[idx * 32 + n] * h + Pq[idx * 32 + 16 + n];
  }
}

// ---------------- scan pass C: replay with real h0; fuse C-dot, D, silu(res) ---
__global__ __launch_bounds__(128)
void k_scan_final(const float* __restrict__ delta, const float* __restrict__ u,
                  const float* __restrict__ dbc, const float* __restrict__ A_log,
                  const float* __restrict__ h0, const float* __restrict__ Dp,
                  const float* __restrict__ xz, unsigned short* __restrict__ ybf) {
  int b = blockIdx.x / NCH, ch = blockIdx.x % NCH;
  int d = threadIdx.x;
  float Ar[DST];
#pragma unroll
  for (int n = 0; n < DST; n++) Ar[n] = -__expf(A_log[d * DST + n]);
  float h[DST];
  size_t hidx = ((size_t)blockIdx.x * DIN + d) * DST;
#pragma unroll
  for (int n = 0; n < DST; n++) h[n] = h0[hidx + n];
  float Dd = Dp[d];
  int l0 = ch * LCH;
  for (int t = 0; t < LCH; t++) {
    size_t m = (size_t)b * LSEQ + l0 + t;
    float de = delta[m * DIN + d];
    float uu = u[m * DIN + d];
    __builtin_prefetch(&delta[(m + 1) * DIN + d], 0, 1);
    __builtin_prefetch(&u[(m + 1) * DIN + d], 0, 1);
    const float* bp = dbc + m * 36 + DTR;
    const float* cp = bp + DST;
    float du = de * uu;
    float y = 0.f;
#pragma unroll
    for (int n = 0; n < DST; n++) {
      float dA = __expf(de * Ar[n]);
      h[n] = dA * h[n] + du * bp[n];
      y += h[n] * cp[n];
    }
    y += uu * Dd;
    float rv = xz[m * (2 * DIN) + DIN + d];      // res half of in_proj output
    y *= rv * sigmoidf_(rv);                      // * silu(res)
    ybf[m * DIN + d] = f2bf_bits(y);
  }
}

// ---------------- final: out = identity + transpose(z) ------------------------
__global__ __launch_bounds__(256)
void k_final(const float* __restrict__ x, const float* __restrict__ z,
             float* __restrict__ out) {
  __shared__ float tile[64][65];
  int b  = blockIdx.x / (LSEQ / 64);
  int lt = blockIdx.x % (LSEQ / 64);
  for (int e = threadIdx.x; e < 64 * 64; e += 256) {
    int li = e >> 6, c = e & 63;                               // coalesced over c
    tile[li][c] = z[((size_t)(b * LSEQ + lt * 64 + li)) * NF_ + c];
  }
  __syncthreads();
  for (int e = threadIdx.x; e < 64 * 64; e += 256) {
    int c = e >> 6, li = e & 63;                               // coalesced over li
    size_t idx = ((size_t)(b * NF_ + c)) * LSEQ + lt * 64 + li;
    out[idx] = x[idx] + tile[li][c];
  }
}

extern "C" void kernel_launch(void* const* d_in, const int* in_sizes, int n_in,
                              void* d_out, int out_size, void* d_ws, size_t ws_size,
                              hipStream_t stream) {
  const float* x         = (const float*)d_in[0];
  const float* conv1_w   = (const float*)d_in[1];
  const float* conv1_b   = (const float*)d_in[2];
  const float* conv2_w   = (const float*)d_in[3];
  const float* conv2_b   = (const float*)d_in[4];
  const float* in_proj_w = (const float*)d_in[5];
  const float* conv1d_w  = (const float*)d_in[6];
  const float* conv1d_b  = (const float*)d_in[7];
  const float* x_proj_w  = (const float*)d_in[8];
  const float* dt_proj_w = (const float*)d_in[9];
  const float* dt_proj_b = (const float*)d_in[10];
  const float* A_log     = (const float*)d_in[11];
  const float* Dp        = (const float*)d_in[12];
  const float* out_proj_w= (const float*)d_in[13];
  float* out = (float*)d_out;
  (void)in_sizes; (void)n_in; (void)out_size; (void)ws_size;

  char* wsp = (char*)d_ws;
  size_t off = 0;
  auto nxt = [&](size_t bytes) -> void* {
    void* p = wsp + off; off += (bytes + 255) & ~(size_t)255; return p;
  };
  unsigned short* xt   = (unsigned short*)nxt((size_t)MTOK * NF_ * 2);
  unsigned short* tok  = (unsigned short*)nxt((size_t)MTOK * NF_ * 2);
  float*          xz   = (float*)         nxt((size_t)MTOK * 2 * DIN * 4);
  float*          xcs  = (float*)         nxt((size_t)MTOK * DIN * 4);
  unsigned short* xcsb = (unsigned short*)nxt((size_t)MTOK * DIN * 2);
  float*          dbc  = (float*)         nxt((size_t)MTOK * 36 * 4);
  float*          dlt  = (float*)         nxt((size_t)MTOK * DIN * 4);
  float*          Pq   = (float*)         nxt((size_t)BN * NCH * DIN * 32 * 4);
  float*          h0   = (float*)         nxt((size_t)BN * NCH * DIN * DST * 4);
  unsigned short* ybf  = (unsigned short*)nxt((size_t)MTOK * DIN * 2);
  unsigned short* otb  = (unsigned short*)nxt((size_t)MTOK * NF_ * 2);
  float*          zf   = (float*)         nxt((size_t)MTOK * NF_ * 4);
  // pre-converted bf16 weights (x_proj zero-padded 36 -> 48 rows)
  unsigned short* w1b  = (unsigned short*)nxt((size_t)64  * 64  * 2);
  unsigned short* wib  = (unsigned short*)nxt((size_t)256 * 64  * 2);
  unsigned short* wxb  = (unsigned short*)nxt((size_t)48  * 128 * 2);
  unsigned short* wob  = (unsigned short*)nxt((size_t)64  * 128 * 2);
  unsigned short* w2b  = (unsigned short*)nxt((size_t)64  * 64  * 2);

  const int EW_BLOCKS = (MTOK * DIN) / 256;              // 18432
  const int TR_BLOCKS = BN * (LSEQ / 64);                // 576
  auto gemm_blocks = [](int M, int Ngroups) { return (((M >> 4) * Ngroups) + 7) / 8; };
  auto cvt_blocks  = [](int n) { return (n + 255) / 256; };

  // 0) weights -> bf16 once (tiny; removes all f32->bf16 work from GEMM loops)
  k_w2bf<<<cvt_blocks(64 * 64), 256, 0, stream>>>(conv1_w, w1b, 64 * 64, 64 * 64);
  k_w2bf<<<cvt_blocks(256 * 64), 256, 0, stream>>>(in_proj_w, wib, 256 * 64, 256 * 64);
  k_w2bf<<<cvt_blocks(48 * 128), 256, 0, stream>>>(x_proj_w, wxb, 36 * 128, 48 * 128);
  k_w2bf<<<cvt_blocks(64 * 128), 256, 0, stream>>>(out_proj_w, wob, 64 * 128, 64 * 128);
  k_w2bf<<<cvt_blocks(64 * 64), 256, 0, stream>>>(conv2_w, w2b, 64 * 64, 64 * 64);

  // 1) x (B,C,L) -> bf16 tokens
  k_x_to_tokens<<<TR_BLOCKS, 256, 0, stream>>>(x, xt);
  // 2) conv1 (1x1): tok = xt @ conv1_w.T + b      (M x 64, K=64) -> bf16
  k_gemm_wmma<true, 4><<<gemm_blocks(MTOK, 1), 256, 0, stream>>>(
      xt, w1b, conv1_b, tok, MTOK, 1, 64, 64, 64);
  // 3) in_proj: xz = tok @ in_proj_w.T            (M x 256, K=64) -> f32
  k_gemm_wmma<false, 4><<<gemm_blocks(MTOK, 4), 256, 0, stream>>>(
      tok, wib, nullptr, xz, MTOK, 4, 256, 64, 256);
  // 4) depthwise causal conv1d + SiLU
  k_dwconv_silu<<<EW_BLOCKS, 256, 0, stream>>>(xz, conv1d_w, conv1d_b, xcs, xcsb);
  // 5) x_proj: dbc = xcs @ x_proj_w.T             (M x 36, K=128, padded to 48)
  k_gemm_wmma<false, 3><<<gemm_blocks(MTOK, 1), 256, 0, stream>>>(
      xcsb, wxb, nullptr, dbc, MTOK, 1, 36, 128, 36);
  // 6) delta = softplus(dt @ dt_proj_w.T + b)
  k_delta<<<EW_BLOCKS, 256, 0, stream>>>(dbc, dt_proj_w, dt_proj_b, dlt);
  // 7-9) chunked parallel selective scan
  k_scan_local<<<BN * NCH, 128, 0, stream>>>(dlt, xcs, dbc, A_log, Pq);
  k_scan_prefix<<<(BN * DIN * DST + 255) / 256, 256, 0, stream>>>(Pq, h0);
  k_scan_final<<<BN * NCH, 128, 0, stream>>>(dlt, xcs, dbc, A_log, h0, Dp, xz, ybf);
  // 10) out_proj: otb = y @ out_proj_w.T          (M x 64, K=128) -> bf16
  k_gemm_wmma<true, 4><<<gemm_blocks(MTOK, 1), 256, 0, stream>>>(
      ybf, wob, nullptr, otb, MTOK, 1, 64, 128, 64);
  // 11) conv2 (1x1): zf = otb @ conv2_w.T + conv2_b
  k_gemm_wmma<false, 4><<<gemm_blocks(MTOK, 1), 256, 0, stream>>>(
      otb, w2b, conv2_b, zf, MTOK, 1, 64, 64, 64);
  // 12) residual add with transpose back to (B,C,H,W)
  k_final<<<TR_BLOCKS, 256, 0, stream>>>(x, zf, out);
}
